// SimplifiedONNXWrapper_6622839570758
// MI455X (gfx1250) — matrix operand order repaired and checked by reading, not compiled, not run on hardware
//
#include <hip/hip_runtime.h>
#include <hip/hip_bf16.h>
#include <math.h>

// ---------------------------------------------------------------------------
// Problem constants (from the reference)
// ---------------------------------------------------------------------------
#define BB    4096
#define TT    1024
#define CC    3
#define FEAT  30
#define HID   64
#define DK    32
#define NC    4

typedef __bf16 bf16_t;
typedef __attribute__((ext_vector_type(16))) __bf16 v16bf;
typedef __attribute__((ext_vector_type(8)))  float  v8f;

// pack two floats into one dword of bf16 pairs (lowers to v_cvt_pk_bf16_f32)
__device__ __forceinline__ unsigned pk_bf16(float a, float b) {
  unsigned short x = __builtin_bit_cast(unsigned short, (bf16_t)a);
  unsigned short y = __builtin_bit_cast(unsigned short, (bf16_t)b);
  return (unsigned)x | ((unsigned)y << 16);
}

// async global -> LDS stage of one contiguous region (bytes % (threads*16)==0)
// each lane issues GLOBAL_LOAD_ASYNC_TO_LDS_B128 (ASYNCcnt-tracked, 16B/lane).
__device__ __forceinline__ void async_stage_row(const void* gsrc, void* lds,
                                                int bytes, int tid, int nthreads) {
  const char* g = (const char*)gsrc;
  unsigned lbase = (unsigned)(unsigned long long)lds;  // low 32b = LDS byte addr
  for (int i = tid * 16; i < bytes; i += nthreads * 16) {
    asm volatile("global_load_async_to_lds_b128 %0, %1, off"
                 :: "v"(lbase + (unsigned)i), "v"(g + i) : "memory");
  }
  asm volatile("s_wait_asynccnt 0x0" ::: "memory");
}

// ---------------------------------------------------------------------------
// Kernel 0: weight prep — reorder conv weights into im2col A-matrix order,
// pad K to 32 (encode/conv1) and rows to 32 (conv2), convert to bf16.
// ---------------------------------------------------------------------------
__global__ void k_prep_weights(const float* __restrict__ Wc_v,
                               const float* __restrict__ Wc_p,
                               const float* __restrict__ W1,
                               const float* __restrict__ W2,
                               bf16_t* __restrict__ Wp_v,
                               bf16_t* __restrict__ Wp_p,
                               bf16_t* __restrict__ W1p,
                               bf16_t* __restrict__ W2p) {
  int tid = threadIdx.x;
  for (int i = tid; i < 64 * 32; i += blockDim.x) {
    int o = i >> 5, j = i & 31;
    float v = 0.f, p = 0.f, w1 = 0.f;
    if (j < 9)  { int k = j / 3, c = j % 3; v  = Wc_v[(o * 3 + c) * 3 + k];
                                            p  = Wc_p[(o * 3 + c) * 3 + k]; }
    if (j < 15) { int k = j / 3, c = j % 3; w1 = W1[(o * 3 + c) * 5 + k]; }
    Wp_v[i] = (bf16_t)v; Wp_p[i] = (bf16_t)p; W1p[i] = (bf16_t)w1;
  }
  for (int i = tid; i < 32 * 320; i += blockDim.x) {
    int o = i / 320, f = i % 320;
    float v = 0.f;
    if (o < FEAT) { int p = f >> 6, ch = f & 63; v = W2[(o * 64 + ch) * 5 + p]; }
    W2p[i] = (bf16_t)v;
  }
}

// ---------------------------------------------------------------------------
// Kernel 1: encode conv (K=3, C=3, 64 out-ch) + ReLU + time-mean -> h[B][64]
// One workgroup (4 waves) per batch row; wave w owns channels 16w..16w+15.
// B-operand = one unconditional aligned 32B LDS load per lane (lanes 16..31
// hit a shared zero row -> no EXEC divergence). Last tile peeled for masking.
// ---------------------------------------------------------------------------
__global__ __launch_bounds__(128)
void k_encode_conv(const float* __restrict__ S, const bf16_t* __restrict__ Wp,
                   const float* __restrict__ bias, float* __restrict__ hout) {
  __shared__ __align__(16) float  xs[TT * CC];      // 12 KB raw row
  __shared__ __align__(64) bf16_t xb[TT * 16];      // 32 KB im2col windows
  __shared__ __align__(64) bf16_t zpad[16];         // zero row for upper half

  const int b = blockIdx.x, tid = threadIdx.x;

  async_stage_row(S + (size_t)b * (TT * CC), xs, TT * CC * 4, tid, 128);
  if (tid < 16) zpad[tid] = (bf16_t)0.f;
  if (b + 1 < BB) __builtin_prefetch(S + (size_t)(b + 1) * (TT * CC), 0, 1);
  __syncthreads();

  // build im2col: position t -> 16 bf16 (9 real taps + zero pad), 32B aligned
  for (int t = tid; t < TT; t += 128) {
    const bool valid = (t < TT - 2);
    float f[9];
#pragma unroll
    for (int e = 0; e < 9; e++) f[e] = valid ? xs[t * 3 + e] : 0.f;
    uint4 lo = make_uint4(pk_bf16(f[0], f[1]), pk_bf16(f[2], f[3]),
                          pk_bf16(f[4], f[5]), pk_bf16(f[6], f[7]));
    uint4 hi = make_uint4(pk_bf16(f[8], 0.f), 0u, 0u, 0u);
    uint4* dst = (uint4*)(xb + t * 16);
    dst[0] = lo; dst[1] = hi;
  }
  __syncthreads();

  const int wave = tid >> 5, lane = tid & 31, col = lane & 15, lh = lane >> 4;
  const int mtile = wave;

  // A-matrix: row = lane%16, contiguous 16-wide K run at 16*lh (this code's k order; B uses the same)
  v16bf a;
  {
    const bf16_t* ar = Wp + (mtile * 16 + col) * 32;
#pragma unroll
    for (int e = 0; e < 8; e++) { a[e] = ar[16 * lh + e]; a[8 + e] = ar[16 * lh + 8 + e]; }
  }
  float brow[8];
  {
    const float* bp = bias + mtile * 16 + 8 * lh;
#pragma unroll
    for (int r = 0; r < 8; r++) brow[r] = bp[r];
  }

  const bf16_t* bsrc = lh ? zpad : xb;       // upper half reads the zero row
  const unsigned amask = lh ? 0u : ~0u;

  float acc[8];
#pragma unroll
  for (int r = 0; r < 8; r++) acc[r] = 0.f;

  for (int nt = 0; nt < 63; nt++) {          // all columns valid (t <= 1007)
    const int t = nt * 16 + col;
    v16bf bv = *(const v16bf*)(bsrc + (((unsigned)(t * 16)) & amask));
    v8f c = {};
    c = __builtin_amdgcn_wmma_f32_16x16x32_bf16(false, a, false, bv,
                                                (short)0, c, false, false);
#pragma unroll
    for (int r = 0; r < 8; r++) acc[r] += fmaxf(c[r] + brow[r], 0.f);
  }
  {                                          // peeled tile 63: mask t >= 1022
    const int t = 63 * 16 + col;
    v16bf bv = *(const v16bf*)(bsrc + (((unsigned)(t * 16)) & amask));
    v8f c = {};
    c = __builtin_amdgcn_wmma_f32_16x16x32_bf16(false, a, false, bv,
                                                (short)0, c, false, false);
    const float vm = (t < TT - 2) ? 1.f : 0.f;
#pragma unroll
    for (int r = 0; r < 8; r++) acc[r] += fmaxf(c[r] + brow[r], 0.f) * vm;
  }

#pragma unroll
  for (int r = 0; r < 8; r++) {
    acc[r] += __shfl_xor(acc[r], 1, 32);
    acc[r] += __shfl_xor(acc[r], 2, 32);
    acc[r] += __shfl_xor(acc[r], 4, 32);
    acc[r] += __shfl_xor(acc[r], 8, 32);
  }
  if (col == 0) {  // lanes 0 and 16
    float* hp = hout + (size_t)b * HID + mtile * 16 + 8 * lh;
#pragma unroll
    for (int r = 0; r < 8; r++) hp[r] = acc[r] * (1.0f / 1022.0f);
  }
}

// ---------------------------------------------------------------------------
// Kernel 2: physical model, fused per batch row:
//   conv1 (K=5,C=3,64ch) -> maxpool4 -> LDS bf16 h1[255][64]
//   conv2 (K=5 over 64ch => GEMM K=320, 10 chained WMMAs) -> pool4 -> mean
// LDS overlay: the raw float row shares memory with h1; total LDS ~ 63.9 KB.
// Clamped columns (t>1019 / t>250) are excluded by pool-group masks.
// ---------------------------------------------------------------------------
__global__ __launch_bounds__(128)
void k_physical(const float* __restrict__ S, const bf16_t* __restrict__ W1p,
                const float* __restrict__ b1, const bf16_t* __restrict__ W2p,
                const float* __restrict__ b2, float* __restrict__ y30) {
  __shared__ __align__(64) unsigned char smem[255 * 64 * 2]; // xs | h1 overlay
  __shared__ __align__(64) bf16_t xb[1020 * 16];             // 32,640 B
  __shared__ __align__(64) bf16_t zpad[16];
  __shared__ float yacc[32];

  float*  xs = (float*)smem;      // phase 0 only
  bf16_t* h1 = (bf16_t*)smem;     // phase 1 output / phase 2 input

  const int b = blockIdx.x, tid = threadIdx.x;

  async_stage_row(S + (size_t)b * (TT * CC), xs, TT * CC * 4, tid, 128);
  if (tid < 16) zpad[tid] = (bf16_t)0.f;
  if (tid < 32) yacc[tid] = 0.f;
  __syncthreads();

  // im2col for conv1: 15 taps + 1 zero per position, 32B aligned rows
  for (int t = tid; t < 1020; t += 128) {
    float f[15];
#pragma unroll
    for (int e = 0; e < 15; e++) f[e] = xs[t * 3 + e];   // max idx 3071
    uint4 lo = make_uint4(pk_bf16(f[0], f[1]),  pk_bf16(f[2], f[3]),
                          pk_bf16(f[4], f[5]),  pk_bf16(f[6], f[7]));
    uint4 hi = make_uint4(pk_bf16(f[8], f[9]),  pk_bf16(f[10], f[11]),
                          pk_bf16(f[12], f[13]), pk_bf16(f[14], 0.f));
    uint4* dst = (uint4*)(xb + t * 16);
    dst[0] = lo; dst[1] = hi;
  }
  __syncthreads();   // xs fully consumed; smem may now be reused as h1

  const int wave = tid >> 5, lane = tid & 31, col = lane & 15, lh = lane >> 4;
  const bf16_t* bsrc = lh ? zpad : xb;
  const unsigned amask = lh ? 0u : ~0u;

  // ---- phase 1: conv1 + pool4 -> h1 -------------------------------------
  {
    const int mtile = wave;
    v16bf a;
    const bf16_t* ar = W1p + (mtile * 16 + col) * 32;
#pragma unroll
    for (int e = 0; e < 8; e++) { a[e] = ar[16 * lh + e]; a[8 + e] = ar[16 * lh + 8 + e]; }
    float brow[8];
    const float* bp = b1 + mtile * 16 + 8 * lh;
#pragma unroll
    for (int r = 0; r < 8; r++) brow[r] = bp[r];

    const bool lead0 = ((col & 3) == 0);
    for (int nt = 0; nt < 63; nt++) {        // t <= 1007, all groups valid
      const int t = nt * 16 + col;
      v16bf bv = *(const v16bf*)(bsrc + (((unsigned)(t * 16)) & amask));
      v8f c = {};
      c = __builtin_amdgcn_wmma_f32_16x16x32_bf16(false, a, false, bv,
                                                  (short)0, c, false, false);
      const int g = nt * 4 + (col >> 2);
      float pv[8];
#pragma unroll
      for (int r = 0; r < 8; r++) {
        float v = c[r] + brow[r];            // no ReLU in conv1
        v = fmaxf(v, __shfl_xor(v, 1, 32));  // maxpool4 over time
        v = fmaxf(v, __shfl_xor(v, 2, 32));
        pv[r] = v;
      }
      if (lead0) {                           // packed 16B store
        uint4 u = make_uint4(pk_bf16(pv[0], pv[1]), pk_bf16(pv[2], pv[3]),
                             pk_bf16(pv[4], pv[5]), pk_bf16(pv[6], pv[7]));
        *(uint4*)(h1 + g * 64 + mtile * 16 + 8 * lh) = u;
      }
    }
    {                                        // peeled tile 63: clamp + g mask
      const int t = 63 * 16 + col;
      const int tc = t < 1020 ? t : 1019;
      v16bf bv = *(const v16bf*)(bsrc + (((unsigned)(tc * 16)) & amask));
      v8f c = {};
      c = __builtin_amdgcn_wmma_f32_16x16x32_bf16(false, a, false, bv,
                                                  (short)0, c, false, false);
      const int g = 63 * 4 + (col >> 2);
      float pv[8];
#pragma unroll
      for (int r = 0; r < 8; r++) {
        float v = c[r] + brow[r];
        v = fmaxf(v, __shfl_xor(v, 1, 32));
        v = fmaxf(v, __shfl_xor(v, 2, 32));
        pv[r] = v;
      }
      if (lead0 && g < 255) {
        uint4 u = make_uint4(pk_bf16(pv[0], pv[1]), pk_bf16(pv[2], pv[3]),
                             pk_bf16(pv[4], pv[5]), pk_bf16(pv[6], pv[7]));
        *(uint4*)(h1 + g * 64 + mtile * 16 + 8 * lh) = u;
      }
    }
  }
  __syncthreads();

  // ---- phase 2: conv2 (GEMM K=320) + pool4 + mean -----------------------
  {
    const int m2 = wave & 1, nhalf = wave >> 1;
    float brow2[8];
#pragma unroll
    for (int r = 0; r < 8; r++) {
      int ch = m2 * 16 + 8 * lh + r;
      brow2[r] = (ch < FEAT) ? b2[ch] : 0.f;
    }
    const bf16_t* a2base = W2p + (m2 * 16 + col) * 320;
    float sacc[8];
#pragma unroll
    for (int r = 0; r < 8; r++) sacc[r] = 0.f;

    for (int nt = nhalf * 8; nt < nhalf * 8 + 8; nt++) {
      const int t = nt * 16 + col;
      const int t2 = t < 251 ? t : 250;                 // clamp (masked later)
      v8f c = {};
      for (int kstep = 0; kstep < 10; kstep++) {        // K = 320
        v16bf a2;
        const bf16_t* ap = a2base + kstep * 32 + 16 * lh;
#pragma unroll
        for (int e = 0; e < 8; e++) { a2[e] = ap[e]; a2[8 + e] = ap[8 + e]; }
        // contiguous im2col row in LDS, 32B aligned
        v16bf bv = *(const v16bf*)(h1 + t2 * 64 + kstep * 32 + 16 * lh);
        c = __builtin_amdgcn_wmma_f32_16x16x32_bf16(false, a2, false, bv,
                                                    (short)0, c, false, false);
      }
      const int g = nt * 4 + (col >> 2);
      const bool lead = ((col & 3) == 0) && (g < 62);
#pragma unroll
      for (int r = 0; r < 8; r++) {
        float v = c[r] + brow2[r];
        v = fmaxf(v, __shfl_xor(v, 1, 32));
        v = fmaxf(v, __shfl_xor(v, 2, 32));
        int ch = m2 * 16 + 8 * lh + r;
        if (lead && ch < FEAT) sacc[r] += v;
      }
    }
#pragma unroll
    for (int r = 0; r < 8; r++) {
      sacc[r] += __shfl_xor(sacc[r], 4, 32);
      sacc[r] += __shfl_xor(sacc[r], 8, 32);
    }
    if (col == 0) {
#pragma unroll
      for (int r = 0; r < 8; r++) {
        int ch = m2 * 16 + 8 * lh + r;
        if (ch < FEAT) atomicAdd(&yacc[ch], sacc[r]);   // ds_add_f32
      }
    }
  }
  __syncthreads();
  if (tid < FEAT) y30[(size_t)b * FEAT + tid] = yacc[tid] * (1.0f / 62.0f);
}

// ---------------------------------------------------------------------------
// Kernel 3: head GEMMs  h[B,64] @ {Ws,Wd}[64,30] + bias   (tiny, VALU)
// ---------------------------------------------------------------------------
__global__ void k_heads1(const float* __restrict__ h_v, const float* __restrict__ h_p,
                         const float* __restrict__ Ws_v, const float* __restrict__ bs_v,
                         const float* __restrict__ Wd_v, const float* __restrict__ bd_v,
                         const float* __restrict__ Ws_p, const float* __restrict__ bs_p,
                         const float* __restrict__ Wd_p, const float* __restrict__ bd_p,
                         float* fv_s, float* fv_d, float* fp_s, float* fp_d) {
  int idx = blockIdx.x * blockDim.x + threadIdx.x;
  if (idx >= BB * FEAT) return;
  int b = idx / FEAT, f = idx % FEAT;
  const float* hv = h_v + (size_t)b * HID;
  const float* hp = h_p + (size_t)b * HID;
  float sv = bs_v[f], dv = bd_v[f], sp = bs_p[f], dp = bd_p[f];
  for (int k = 0; k < HID; k++) {
    sv += hv[k] * Ws_v[k * FEAT + f];
    dv += hv[k] * Wd_v[k * FEAT + f];
    sp += hp[k] * Ws_p[k * FEAT + f];
    dp += hp[k] * Wd_p[k * FEAT + f];
  }
  fv_s[idx] = sv; fv_d[idx] = dv; fp_s[idx] = sp; fp_d[idx] = dp;
}

// shared-space projection: f*_sp = f*_s @ Wsp + bsp
__global__ void k_heads2(const float* __restrict__ fv_s, const float* __restrict__ fp_s,
                         const float* __restrict__ Wsp, const float* __restrict__ bsp,
                         float* fv_sp, float* fp_sp) {
  int idx = blockIdx.x * blockDim.x + threadIdx.x;
  if (idx >= BB * FEAT) return;
  int b = idx / FEAT, f = idx % FEAT;
  const float* av = fv_s + (size_t)b * FEAT;
  const float* ap = fp_s + (size_t)b * FEAT;
  float sv = bsp[f], sp = bsp[f];
  for (int k = 0; k < FEAT; k++) {
    float w = Wsp[k * FEAT + f];
    sv += av[k] * w;
    sp += ap[k] * w;
  }
  fv_sp[idx] = sv; fp_sp[idx] = sp;
}

// ---------------------------------------------------------------------------
// Kernel 4: logsumexp soft-OR fusion + masked 4-token multi-query attention.
// One wave per batch element (4 waves / block).
// ---------------------------------------------------------------------------
__global__ __launch_bounds__(128)
void k_fuse_attn(const int* __restrict__ pairs,
                 const float* __restrict__ fv_sp, const float* __restrict__ fp_sp,
                 const float* __restrict__ fv_d,  const float* __restrict__ fp_d,
                 const float* __restrict__ y30,
                 const float* __restrict__ Wa, const float* __restrict__ ba,
                 const float* __restrict__ Wf, const float* __restrict__ bf_,
                 float* __restrict__ out) {
  __shared__ float tok[4][4][FEAT];
  __shared__ float qk[4][4][(2 + NC) * DK];
  __shared__ float qm[4][DK];
  __shared__ float wm[4][NC * 4];

  const int tid = threadIdx.x, w = tid >> 5, lane = tid & 31;
  const int b = blockIdx.x * 4 + w;
  const int pr = pairs[b];
  const float pf = (float)pr;

  if (lane < FEAT) {
    size_t o = (size_t)b * FEAT + lane;
    float h1 = fv_sp[o], h2 = fp_sp[o], h12 = h1 + h2;
    float m1 = fmaxf(h12, fmaxf(h1, h2));
    float lse1 = m1 + logf(2.f * expf(h12 - m1) + expf(h1 - m1) + expf(h2 - m1));
    float m2 = fmaxf(0.f, fmaxf(h1, h2));
    float lse2 = m2 + logf(2.f * expf(-m2) + expf(h1 - m2) + expf(h2 - m2));
    float fa = pf * (lse1 - lse2) + (1.f - pf) * h2;
    tok[w][0][lane] = fp_d[o];
    tok[w][1][lane] = fa;
    tok[w][2][lane] = y30[o];
    tok[w][3][lane] = fv_d[o];
  }
  __syncthreads();

  for (int i = 0; i < 6; i++) {
    int j = 32 * i + lane;
    float s0 = ba[j], s1 = ba[j], s2 = ba[j], s3 = ba[j];
    for (int k = 0; k < FEAT; k++) {
      float wv = Wa[k * (2 + NC) * DK + j];
      s0 += tok[w][0][k] * wv;
      s1 += tok[w][1][k] * wv;
      s2 += tok[w][2][k] * wv;
      s3 += tok[w][3][k] * wv;
    }
    qk[w][0][j] = s0; qk[w][1][j] = s1; qk[w][2][j] = s2; qk[w][3][j] = s3;
  }
  __syncthreads();

  {
    float q0 = qk[w][0][lane], q1 = qk[w][1][lane],
          q2 = qk[w][2][lane], q3 = qk[w][3][lane];
    qm[w][lane] = pf * 0.25f * (q0 + q1 + q2 + q3) +
                  (1.f - pf) * (1.f / 3.f) * (q0 + q1 + q2);
  }
  __syncthreads();

  if (lane < NC * 4) {
    int n = lane >> 2, t = lane & 3;
    float lg = 0.f;
    for (int d = 0; d < DK; d++) lg += qk[w][t][2 * DK + n * DK + d] * qm[w][d];
    lg *= 0.17677669529663687f;               // 1/sqrt(DK)
    if (t == 3 && pr == 0) lg = -1e30f;
    float mx = fmaxf(lg, __shfl_xor(lg, 1, 32));
    mx = fmaxf(mx, __shfl_xor(mx, 2, 32));
    float e = expf(lg - mx);
    float s = e + __shfl_xor(e, 1, 32);
    s += __shfl_xor(s, 2, 32);
    wm[w][lane] = e / s;
  }
  __syncthreads();

  for (int n = 0; n < NC; n++) {
    float ff = 0.f;
    for (int t = 0; t < 4; t++) ff += wm[w][n * 4 + t] * qk[w][t][DK + lane];
    float contrib = ff * Wf[lane * NC + n];
    contrib += __shfl_xor(contrib, 1, 32);
    contrib += __shfl_xor(contrib, 2, 32);
    contrib += __shfl_xor(contrib, 4, 32);
    contrib += __shfl_xor(contrib, 8, 32);
    contrib += __shfl_xor(contrib, 16, 32);
    if (lane == 0) out[(size_t)b * NC + n] = contrib + bf_[n];
  }
}

// ---------------------------------------------------------------------------
// Launch
// ---------------------------------------------------------------------------
extern "C" void kernel_launch(void* const* d_in, const int* in_sizes, int n_in,
                              void* d_out, int out_size, void* d_ws, size_t ws_size,
                              hipStream_t stream) {
  const int*   pairs = (const int*)  d_in[0];
  const float* S_V   = (const float*)d_in[1];
  const float* S_P   = (const float*)d_in[2];
  const float* S_P1  = (const float*)d_in[3];
  const float* Wc_v  = (const float*)d_in[4];
  const float* bc_v  = (const float*)d_in[5];
  const float* Ws_v  = (const float*)d_in[6];
  const float* bs_v  = (const float*)d_in[7];
  const float* Wd_v  = (const float*)d_in[8];
  const float* bd_v  = (const float*)d_in[9];
  const float* Wc_p  = (const float*)d_in[10];
  const float* bc_p  = (const float*)d_in[11];
  const float* Ws_p  = (const float*)d_in[12];
  const float* bs_p  = (const float*)d_in[13];
  const float* Wd_p  = (const float*)d_in[14];
  const float* bd_p  = (const float*)d_in[15];
  const float* Wsp   = (const float*)d_in[16];
  const float* bsp   = (const float*)d_in[17];
  const float* W1    = (const float*)d_in[18];
  const float* b1    = (const float*)d_in[19];
  const float* W2    = (const float*)d_in[20];
  const float* b2    = (const float*)d_in[21];
  const float* Wa    = (const float*)d_in[22];
  const float* ba    = (const float*)d_in[23];
  const float* Wf    = (const float*)d_in[24];
  const float* bf_   = (const float*)d_in[25];
  float* out = (float*)d_out;

  char* w = (char*)d_ws;
  size_t off = 0;
  auto carve = [&](size_t bytes) { char* p = w + off; off = (off + bytes + 255) & ~(size_t)255; return p; };
  bf16_t* Wp_v = (bf16_t*)carve(64 * 32 * sizeof(bf16_t));
  bf16_t* Wp_p = (bf16_t*)carve(64 * 32 * sizeof(bf16_t));
  bf16_t* W1p  = (bf16_t*)carve(64 * 32 * sizeof(bf16_t));
  bf16_t* W2p  = (bf16_t*)carve(32 * 320 * sizeof(bf16_t));
  float* h_v   = (float*)carve((size_t)BB * HID * sizeof(float));
  float* h_p   = (float*)carve((size_t)BB * HID * sizeof(float));
  float* fv_s  = (float*)carve((size_t)BB * FEAT * sizeof(float));
  float* fp_s  = (float*)carve((size_t)BB * FEAT * sizeof(float));
  float* fv_d  = (float*)carve((size_t)BB * FEAT * sizeof(float));
  float* fp_d  = (float*)carve((size_t)BB * FEAT * sizeof(float));
  float* fv_sp = (float*)carve((size_t)BB * FEAT * sizeof(float));
  float* fp_sp = (float*)carve((size_t)BB * FEAT * sizeof(float));
  float* y30   = (float*)carve((size_t)BB * FEAT * sizeof(float));

  k_prep_weights<<<1, 256, 0, stream>>>(Wc_v, Wc_p, W1, W2, Wp_v, Wp_p, W1p, W2p);

  k_encode_conv<<<BB, 128, 0, stream>>>(S_V, Wp_v, bc_v, h_v);
  k_encode_conv<<<BB, 128, 0, stream>>>(S_P, Wp_p, bc_p, h_p);
  k_physical  <<<BB, 128, 0, stream>>>(S_P1, W1p, b1, W2p, b2, y30);

  int nh = BB * FEAT;
  k_heads1<<<(nh + 255) / 256, 256, 0, stream>>>(h_v, h_p, Ws_v, bs_v, Wd_v, bd_v,
                                                 Ws_p, bs_p, Wd_p, bd_p,
                                                 fv_s, fv_d, fp_s, fp_d);
  k_heads2<<<(nh + 255) / 256, 256, 0, stream>>>(fv_s, fp_s, Wsp, bsp, fv_sp, fp_sp);

  k_fuse_attn<<<BB / 4, 128, 0, stream>>>(pairs, fv_sp, fp_sp, fv_d, fp_d, y30,
                                          Wa, ba, Wf, bf_, out);
}
